// Attention_48687749267843
// MI455X (gfx1250) — matrix operand order repaired and checked by reading, not compiled, not run on hardware
//
#include <hip/hip_runtime.h>
#include <hip/hip_bf16.h>

typedef __attribute__((ext_vector_type(16))) _Float16 v16h;
typedef __attribute__((ext_vector_type(8)))  float    v8f;

#define BATCH 8
#define DIMC  384           // DIM == MID
#define NH    12
#define HD    32
#define NTOK  1024
#define SCALE 0.17677669529663687f   // 1/sqrt(32)

union U16 { uint4 q[2]; v16h v; };

// A fragment: 16x32 (MxK) f16, row-major source with leading dim `ld` (halves).
// ISA layout: lane&15 = row M; lanes 0-15 hold K 0-7 & 16-23, lanes 16-31 hold K 8-15 & 24-31.
static __device__ inline v16h load_a16(const _Float16* p, int ld, int lane) {
    const _Float16* row = p + (lane & 15) * ld + ((lane >> 4) << 3);
    U16 u;
    u.q[0] = *(const uint4*)(row);
    u.q[1] = *(const uint4*)(row + 16);
    return u.v;
}

// B fragment: 32x16 (KxN) f16 supplied as Bt (N x K row-major, leading dim `ld`).
// lane&15 = column N; lanes 0-15 hold K 0-7 & 16-23, lanes 16-31 hold K 8-15 & 24-31.
static __device__ inline v16h load_b16(const _Float16* p, int ld, int lane) {
    const _Float16* row = p + (lane & 15) * ld + ((lane >> 4) << 3);
    U16 u;
    u.q[0] = *(const uint4*)(row);
    u.q[1] = *(const uint4*)(row + 16);
    return u.v;
}

static __device__ inline v8f wmma_f16(v16h a, v16h b, v8f c) {
    return __builtin_amdgcn_wmma_f32_16x16x32_f16(false, a, false, b, (short)0, c, false, false);
}

// CDNA5 async cache->LDS DMA (no VGPR staging), tracked by ASYNCcnt.
// vdst VGPR holds the 32-bit LDS byte address (= low 32 bits of the generic LDS pointer).
static __device__ inline void async_copy_b128(void* lds_dst, const void* gsrc) {
    unsigned lds_addr = (unsigned)(unsigned long long)lds_dst;
    asm volatile("global_load_async_to_lds_b128 %0, %1, off"
                 :: "v"(lds_addr), "v"(gsrc) : "memory");
}
static __device__ inline void wait_async0() {
    asm volatile("s_wait_asynccnt 0x0" ::: "memory");
}

// ---------------- prep: fp32 -> f16 conversions ----------------
__global__ __launch_bounds__(256) void prep_x_kernel(const float* __restrict__ x,
                                                     _Float16* __restrict__ xt) {
    int idx = blockIdx.x * 256 + threadIdx.x;          // B*DIMC*NTOK = 3145728
    int b   = idx / (DIMC * NTOK);
    int rem = idx % (DIMC * NTOK);
    int c   = rem / NTOK;
    int n   = rem % NTOK;
    xt[(size_t)b * NTOK * DIMC + (size_t)n * DIMC + c] = (_Float16)x[idx];
}

__global__ __launch_bounds__(256) void prep_w_kernel(const float* __restrict__ wq,
                                                     const float* __restrict__ wkv,
                                                     const float* __restrict__ wproj,
                                                     _Float16* __restrict__ wqkv_h,
                                                     _Float16* __restrict__ wproj_h) {
    int idx = blockIdx.x * 256 + threadIdx.x;          // total 589824
    if (idx < 147456)        wqkv_h[idx] = (_Float16)wq[idx];
    else if (idx < 442368)   wqkv_h[idx] = (_Float16)wkv[idx - 147456];
    else                     wproj_h[idx - 442368] = (_Float16)wproj[idx - 442368];
}

// ---------------- QKV projection GEMM + bias + scatter ----------------
// grid (32, 9, 8), block 128. Each wave: one 32x32 tile (2x2 WMMA, A/B frags reused).
__global__ __launch_bounds__(128) void qkv_kernel(const _Float16* __restrict__ xt,
                                                  const _Float16* __restrict__ wqkv,
                                                  const float* __restrict__ bq,
                                                  const float* __restrict__ bkv,
                                                  _Float16* __restrict__ q_h,
                                                  _Float16* __restrict__ k_h,
                                                  _Float16* __restrict__ vt_h) {
    int lane = threadIdx.x & 31, wave = threadIdx.x >> 5;
    int o0 = (blockIdx.y * 4 + wave) * 32;             // 0..1120
    int n0 = blockIdx.x * 32;                          // 0..992
    int b  = blockIdx.z;
    const _Float16* A  = wqkv + (size_t)o0 * DIMC;
    const _Float16* Bt = xt + ((size_t)b * NTOK + n0) * DIMC;
    v8f acc[2][2] = {};
#pragma unroll
    for (int kt = 0; kt < DIMC; kt += 32) {
        v16h a0 = load_a16(A + kt, DIMC, lane);
        v16h a1 = load_a16(A + 16 * DIMC + kt, DIMC, lane);
        v16h b0 = load_b16(Bt + kt, DIMC, lane);
        v16h b1 = load_b16(Bt + 16 * DIMC + kt, DIMC, lane);
        acc[0][0] = wmma_f16(a0, b0, acc[0][0]);
        acc[0][1] = wmma_f16(a0, b1, acc[0][1]);
        acc[1][0] = wmma_f16(a1, b0, acc[1][0]);
        acc[1][1] = wmma_f16(a1, b1, acc[1][1]);
    }
    int n = lane & 15;
    int base8 = (lane >> 4) * 8;
#pragma unroll
    for (int im = 0; im < 2; ++im)
#pragma unroll
    for (int jn = 0; jn < 2; ++jn)
#pragma unroll
    for (int r = 0; r < 8; ++r) {
        int o  = o0 + im * 16 + base8 + r;
        int nn = n0 + jn * 16 + n;
        float val = acc[im][jn][r];
        if (o < DIMC) {                                // Q: scale folded in
            val = (val + bq[o]) * SCALE;
            int h = o >> 5, dd = o & 31;
            q_h[(((size_t)b * NH + h) * NTOK + nn) * HD + dd] = (_Float16)val;
        } else if (o < 2 * DIMC) {                     // K: stored N x d (== Bt of K^T)
            int oc = o - DIMC;
            val += bkv[oc];
            int h = oc >> 5, dd = oc & 31;
            k_h[(((size_t)b * NH + h) * NTOK + nn) * HD + dd] = (_Float16)val;
        } else {                                       // V: stored d x N (V^T == Bt of V)
            int oc = o - 2 * DIMC;
            val += bkv[oc + DIMC];
            int h = oc >> 5, dd = oc & 31;
            vt_h[(((size_t)b * NH + h) * HD + dd) * NTOK + nn] = (_Float16)val;
        }
    }
}

// ---------------- flash attention per (b,h), 128 query rows / block ----------------
// grid (8, 1, 96), block 256, dynamic LDS: K(64KB) + Vt(64KB) + P-bounce(8KB)
__global__ __launch_bounds__(256) void attn_kernel(const _Float16* __restrict__ q_h,
                                                   const _Float16* __restrict__ k_h,
                                                   const _Float16* __restrict__ vt_h,
                                                   const float* __restrict__ bias_table,
                                                   const int* __restrict__ rel_index,
                                                   _Float16* __restrict__ ao) {
    extern __shared__ char smem[];
    _Float16* k_lds = (_Float16*)smem;                 // NTOK*HD
    _Float16* v_lds = k_lds + NTOK * HD;               // NTOK*HD
    _Float16* p_lds = v_lds + NTOK * HD;               // 8 waves * 16*32

    int tid = threadIdx.x;
    int lane = tid & 31, wave = tid >> 5;
    int bh = blockIdx.z;
    int h  = bh % NH;

    // stage K (NxHD) and Vt (HDxN) into LDS via async DMA (no VGPR staging)
    {
        const uint4* ks = (const uint4*)(k_h  + (size_t)bh * NTOK * HD);
        const uint4* vs = (const uint4*)(vt_h + (size_t)bh * NTOK * HD);
        uint4* kd = (uint4*)k_lds;
        uint4* vd = (uint4*)v_lds;
        for (int i = tid; i < NTOK * HD / 8; i += 256) {
            async_copy_b128(kd + i, ks + i);
            async_copy_b128(vd + i, vs + i);
        }
        wait_async0();
    }
    __syncthreads();

    int q0 = (blockIdx.x * 8 + wave) * 16;
    int n = lane & 15;
    int base8 = (lane >> 4) * 8;
    _Float16* pw = p_lds + wave * (16 * 32);

    v16h aq = load_a16(q_h + ((size_t)bh * NTOK + q0) * HD, HD, lane);
    v8f accA = {}, accB = {};                          // output cols 0-15 / 16-31
    float mrow[8], lrow[8];
#pragma unroll
    for (int r = 0; r < 8; ++r) { mrow[r] = -1e30f; lrow[r] = 0.f; }

    for (int kt = 0; kt < NTOK; kt += 32) {
        v8f z = {};
        v16h bk0 = load_b16(k_lds + (size_t)kt * HD, HD, lane);
        v16h bk1 = load_b16(k_lds + (size_t)(kt + 16) * HD, HD, lane);
        v8f s0 = wmma_f16(aq, bk0, z);
        v8f s1 = wmma_f16(aq, bk1, z);

        // relative position bias (gather, L2-resident)
#pragma unroll
        for (int r = 0; r < 8; ++r) {
            int qrow = q0 + base8 + r;
            int i0 = rel_index[qrow * NTOK + kt + n];
            int i1 = rel_index[qrow * NTOK + kt + 16 + n];
            s0[r] += bias_table[i0 * NH + h];
            s1[r] += bias_table[i1 * NH + h];
        }

        // row max over the 32 new columns (reduce across 16 column lanes)
        float tm[8];
#pragma unroll
        for (int r = 0; r < 8; ++r) tm[r] = fmaxf(s0[r], s1[r]);
#pragma unroll
        for (int m = 8; m >= 1; m >>= 1)
#pragma unroll
            for (int r = 0; r < 8; ++r) tm[r] = fmaxf(tm[r], __shfl_xor(tm[r], m, 32));

        // online softmax update; stash P tile (16x32) to LDS for A-layout reload
#pragma unroll
        for (int r = 0; r < 8; ++r) {
            float mnew = fmaxf(mrow[r], tm[r]);
            float corr = __expf(mrow[r] - mnew);
            float p0 = __expf(s0[r] - mnew);
            float p1 = __expf(s1[r] - mnew);
            mrow[r] = mnew;
            lrow[r] = lrow[r] * corr + p0 + p1;        // lane-partial (2 cols), reduced later
            accA[r] *= corr;
            accB[r] *= corr;
            pw[(base8 + r) * 32 + n]      = (_Float16)p0;
            pw[(base8 + r) * 32 + 16 + n] = (_Float16)p1;
        }

        // O += P (16x32) x V (32xHD); Vt rows are HD, contiguous along keys
        v16h ap  = load_a16(pw, 32, lane);
        v16h bv0 = load_b16(v_lds + kt, NTOK, lane);
        v16h bv1 = load_b16(v_lds + 16 * NTOK + kt, NTOK, lane);
        accA = wmma_f16(ap, bv0, accA);
        accB = wmma_f16(ap, bv1, accB);
    }

    // complete the row sums across the 16 column lanes
#pragma unroll
    for (int m = 8; m >= 1; m >>= 1)
#pragma unroll
        for (int r = 0; r < 8; ++r) lrow[r] += __shfl_xor(lrow[r], m, 32);

    // write attention output transposed: ao[b][token][h*32+dd]  (Bt for projection)
    int b = bh / NH;
    _Float16* aob = ao + (size_t)b * NTOK * DIMC;
#pragma unroll
    for (int r = 0; r < 8; ++r) {
        int qrow = q0 + base8 + r;
        float inv = 1.0f / lrow[r];
        aob[(size_t)qrow * DIMC + h * HD + n]      = (_Float16)(accA[r] * inv);
        aob[(size_t)qrow * DIMC + h * HD + 16 + n] = (_Float16)(accB[r] * inv);
    }
}

// ---------------- output projection GEMM + bias -> fp32 ----------------
// grid (32, 3, 8), block 128. Each wave: one 32x32 tile (2x2 WMMA).
__global__ __launch_bounds__(128) void proj_kernel(const _Float16* __restrict__ ao,
                                                   const _Float16* __restrict__ wproj_h,
                                                   const float* __restrict__ bproj,
                                                   float* __restrict__ out) {
    int lane = threadIdx.x & 31, wave = threadIdx.x >> 5;
    int o0 = (blockIdx.y * 4 + wave) * 32;             // 0..352
    int n0 = blockIdx.x * 32;
    int b  = blockIdx.z;
    const _Float16* A  = wproj_h + (size_t)o0 * DIMC;
    const _Float16* Bt = ao + ((size_t)b * NTOK + n0) * DIMC;
    v8f acc[2][2] = {};
#pragma unroll
    for (int kt = 0; kt < DIMC; kt += 32) {
        v16h a0 = load_a16(A + kt, DIMC, lane);
        v16h a1 = load_a16(A + 16 * DIMC + kt, DIMC, lane);
        v16h b0 = load_b16(Bt + kt, DIMC, lane);
        v16h b1 = load_b16(Bt + 16 * DIMC + kt, DIMC, lane);
        acc[0][0] = wmma_f16(a0, b0, acc[0][0]);
        acc[0][1] = wmma_f16(a0, b1, acc[0][1]);
        acc[1][0] = wmma_f16(a1, b0, acc[1][0]);
        acc[1][1] = wmma_f16(a1, b1, acc[1][1]);
    }
    int n = lane & 15;
    int base8 = (lane >> 4) * 8;
#pragma unroll
    for (int im = 0; im < 2; ++im)
#pragma unroll
    for (int jn = 0; jn < 2; ++jn)
#pragma unroll
    for (int r = 0; r < 8; ++r) {
        int o  = o0 + im * 16 + base8 + r;
        int nn = n0 + jn * 16 + n;
        out[((size_t)b * DIMC + o) * NTOK + nn] = acc[im][jn][r] + bproj[o];
    }
}

extern "C" void kernel_launch(void* const* d_in, const int* in_sizes, int n_in,
                              void* d_out, int out_size, void* d_ws, size_t ws_size,
                              hipStream_t stream) {
    const float* x          = (const float*)d_in[0];
    const float* wq         = (const float*)d_in[1];
    const float* bq         = (const float*)d_in[2];
    const float* wkv        = (const float*)d_in[3];
    const float* bkv        = (const float*)d_in[4];
    const float* wproj      = (const float*)d_in[5];
    const float* bproj      = (const float*)d_in[6];
    const float* bias_table = (const float*)d_in[7];
    const int*   rel_index  = (const int*)d_in[8];
    float* out = (float*)d_out;

    // workspace partition (halves)
    _Float16* ws      = (_Float16*)d_ws;
    const size_t XT   = (size_t)BATCH * NTOK * DIMC;   // 3145728
    _Float16* xt      = ws;
    _Float16* wqkv_h  = xt + XT;                       // 1152*384 = 442368
    _Float16* wproj_h = wqkv_h + 442368;               // 147456
    _Float16* q_h     = wproj_h + 147456;
    _Float16* k_h     = q_h + XT;
    _Float16* vt_h    = k_h + XT;
    _Float16* ao      = vt_h + XT;                     // + XT  -> ~32.6 MB total

    prep_x_kernel<<<12288, 256, 0, stream>>>(x, xt);
    prep_w_kernel<<<2304, 256, 0, stream>>>(wq, wkv, wproj, wqkv_h, wproj_h);
    qkv_kernel<<<dim3(32, 9, 8), 128, 0, stream>>>(xt, wqkv_h, bq, bkv, q_h, k_h, vt_h);
    size_t smem = (size_t)(2 * NTOK * HD + 8 * 16 * 32) * sizeof(_Float16); // 139264 B
    attn_kernel<<<dim3(8, 1, 96), 256, smem, stream>>>(q_h, k_h, vt_h, bias_table, rel_index, ao);
    proj_kernel<<<dim3(32, 3, 8), 128, 0, stream>>>(ao, wproj_h, bproj, out);
}
